// NaiveAttention_19731079758126
// MI455X (gfx1250) — matrix operand order repaired and checked by reading, not compiled, not run on hardware
//
#include <hip/hip_runtime.h>
#include <hip/hip_bf16.h>

typedef __bf16 bf16_t;
typedef __attribute__((ext_vector_type(8)))  __bf16 v8bf;
typedef __attribute__((ext_vector_type(16))) __bf16 v16bf;
typedef __attribute__((ext_vector_type(8)))  float  v8f;
typedef int v4i_vs __attribute__((vector_size(16)));   // matches builtin param type

#define D_MODEL 1024
#define SEQ     2048
#define BATCH   2
#define NHEAD   16
#define HD      64
#define MTOT    (BATCH * SEQ)   // 4096

#if defined(__gfx1250__) && __has_builtin(__builtin_amdgcn_global_load_async_to_lds_b128) && \
    __has_builtin(__builtin_amdgcn_s_wait_asynccnt)
#define USE_ASYNC 1
#else
#define USE_ASYNC 0
#endif

__device__ inline v8f v8f_zero() {
  v8f z = {0.f, 0.f, 0.f, 0.f, 0.f, 0.f, 0.f, 0.f};
  return z;
}

// 16-byte global -> LDS copy; async (direct-to-LDS, ASYNCcnt) when available.
__device__ inline void g2l_16B(const bf16_t* g, bf16_t* l) {
#if USE_ASYNC
  __builtin_amdgcn_global_load_async_to_lds_b128(
      (__attribute__((address_space(1))) v4i_vs*)(uintptr_t)g,
      (__attribute__((address_space(3))) v4i_vs*)(unsigned)(uintptr_t)l,
      0, 0);
#else
  *(v8bf*)l = *(const v8bf*)g;
#endif
}

__device__ inline void wait_async0() {
#if USE_ASYNC
  __builtin_amdgcn_s_wait_asynccnt(0);
#endif
}

// ---------------------------------------------------------------- conversion
__global__ void cvt_f32_bf16(const float* __restrict__ in, bf16_t* __restrict__ out, int n) {
  int i = (blockIdx.x * blockDim.x + threadIdx.x) * 4;
  if (i + 3 < n) {
    float4 f = *(const float4*)(in + i);
    out[i + 0] = (bf16_t)f.x;
    out[i + 1] = (bf16_t)f.y;
    out[i + 2] = (bf16_t)f.z;
    out[i + 3] = (bf16_t)f.w;
  }
}

// ---------------------------------------------------------------- WMMA GEMM
// C[M,N] = A[M,K] * W[N,K]^T, bf16 in, fp32 accumulate.
// Async double-buffered LDS staging: loads for step k+1 overlap wmmas of step k.
// MODE 0: write fp32 C row-major to outF.
// MODE 1: QKV epilogue -> scatter q (scaled), k [B,H,T,hd], v^T [B,H,hd,T].
template <int MODE>
__global__ __launch_bounds__(256) void gemm_wmma(
    const bf16_t* __restrict__ A, const bf16_t* __restrict__ W,
    float* __restrict__ outF,
    bf16_t* __restrict__ qb, bf16_t* __restrict__ kbuf, bf16_t* __restrict__ vtb,
    int M, int N, int K) {
  constexpr int LDSW = 40;  // 32 + 8 halfs pad (80B row, 16B aligned)
  __shared__ bf16_t As[2][128 * LDSW];
  __shared__ bf16_t Ws[2][128 * LDSW];

  const int tid  = threadIdx.x;
  const int lane = tid & 31;
  const int wave = tid >> 5;
  const int m0 = blockIdx.y * 128;
  const int n0 = blockIdx.x * 128;
  const int wm = wave >> 1;   // 4 waves along M
  const int wn = wave & 1;    // 2 waves along N
  const int ml = lane & 15;
  const int sh = lane >> 4;

  v8f acc[2][4];
#pragma unroll
  for (int i = 0; i < 2; ++i)
#pragma unroll
    for (int j = 0; j < 4; ++j) acc[i][j] = v8f_zero();

  const int rrow  = tid >> 1;
  const int rhalf = tid & 1;

  // stage one 128x32 A tile + 128x32 W tile into LDS buffer `buf`
  auto stage = [&](int buf, int kb0) {
    const bf16_t* ga = A + (size_t)(m0 + rrow) * K + kb0 + rhalf * 16;
    const bf16_t* gw = W + (size_t)(n0 + rrow) * K + kb0 + rhalf * 16;
    bf16_t* la = &As[buf][rrow * LDSW + rhalf * 16];
    bf16_t* lw = &Ws[buf][rrow * LDSW + rhalf * 16];
    g2l_16B(ga,     la);
    g2l_16B(ga + 8, la + 8);
    g2l_16B(gw,     lw);
    g2l_16B(gw + 8, lw + 8);
  };

  stage(0, 0);
  wait_async0();
  __syncthreads();

  int buf = 0;
  for (int kb0 = 0; kb0 < K; kb0 += 32) {
    if (kb0 + 32 < K) stage(buf ^ 1, kb0 + 32);  // overlap with compute below

    // B fragments: lane n holds 16 contiguous K of W row (column of B)
    v16bf bfrag[4];
#pragma unroll
    for (int nt = 0; nt < 4; ++nt) {
      const bf16_t* src = &Ws[buf][(wn * 64 + nt * 16 + ml) * LDSW + 16 * sh];
      v8bf* p = (v8bf*)&bfrag[nt];
      p[0] = *(const v8bf*)(src);
      p[1] = *(const v8bf*)(src + 8);
    }
#pragma unroll
    for (int mt = 0; mt < 2; ++mt) {
      // A fragment: lane m holds K {16s..16s+15}
      v16bf afrag;
      const bf16_t* src = &As[buf][(wm * 32 + mt * 16 + ml) * LDSW];
      v8bf* p = (v8bf*)&afrag;
      p[0] = *(const v8bf*)(src + 16 * sh);
      p[1] = *(const v8bf*)(src + 16 * sh + 8);
#pragma unroll
      for (int nt = 0; nt < 4; ++nt)
        acc[mt][nt] = __builtin_amdgcn_wmma_f32_16x16x32_bf16(
            false, afrag, false, bfrag[nt], (short)0, acc[mt][nt], false, false);
    }

    wait_async0();     // our own next-tile async writes are done
    __syncthreads();   // everyone else's too
    buf ^= 1;
  }

  // epilogue: C layout = VGPR j -> row (j + 8*sh), lane&15 -> col
#pragma unroll
  for (int mt = 0; mt < 2; ++mt) {
#pragma unroll
    for (int nt = 0; nt < 4; ++nt) {
#pragma unroll
      for (int j = 0; j < 8; ++j) {
        const int row = m0 + wm * 32 + mt * 16 + j + 8 * sh;
        const int col = n0 + wn * 64 + nt * 16 + ml;
        float val = acc[mt][nt][j];
        if (MODE == 0) {
          outF[(size_t)row * N + col] = val;
        } else {
          const int b     = row >> 11;   // row / SEQ
          const int tt    = row & 2047;
          const int which = col >> 10;   // 0=q 1=k 2=v
          const int f     = col & 1023;
          const int h     = f >> 6;
          const int c     = f & 63;
          const int bh    = b * NHEAD + h;
          if (which == 0) {
            qb[((size_t)bh * SEQ + tt) * HD + c] = (bf16_t)(val * 0.125f);  // 1/sqrt(64)
          } else if (which == 1) {
            kbuf[((size_t)bh * SEQ + tt) * HD + c] = (bf16_t)val;
          } else {
            vtb[((size_t)bh * HD + c) * SEQ + tt] = (bf16_t)val;  // transposed
          }
        }
      }
    }
  }
}

// ------------------------------------------------------------ flash attention
// One wave = 16 query rows. S(16x32) via 4 wmmas, online softmax in regs,
// P bounced through per-wave LDS (C-layout -> A-layout), P*V via 4 wmmas.
__global__ __launch_bounds__(128) void flash_attn(
    const bf16_t* __restrict__ q, const bf16_t* __restrict__ k,
    const bf16_t* __restrict__ vt, bf16_t* __restrict__ attnb) {
  constexpr int PSTR = 40;
  __shared__ bf16_t Pl[4][16 * PSTR];

  const int blk  = blockIdx.x;
  const int q64  = blk & 31;   // query block of 64 within T
  const int bh   = blk >> 5;   // b*NHEAD + h
  const int wave = threadIdx.x >> 5;
  const int lane = threadIdx.x & 31;
  const int ml   = lane & 15;
  const int sh   = lane >> 4;
  const int q0   = q64 * 64 + wave * 16;

  const bf16_t* qp = q  + (size_t)bh * SEQ * HD;
  const bf16_t* kp = k  + (size_t)bh * SEQ * HD;
  const bf16_t* vp = vt + (size_t)bh * HD * SEQ;

  // Q fragments for the two K=32 halves of hd=64 (q pre-scaled by 1/sqrt(hd))
  v16bf aq[2];
#pragma unroll
  for (int kk = 0; kk < 2; ++kk) {
    const bf16_t* src = qp + (size_t)(q0 + ml) * HD + kk * 32;
    v8bf* p = (v8bf*)&aq[kk];
    p[0] = *(const v8bf*)(src + 16 * sh);
    p[1] = *(const v8bf*)(src + 16 * sh + 8);
  }

  v8f O[4];
  float mrow[8], lrow[8];
#pragma unroll
  for (int nt = 0; nt < 4; ++nt) O[nt] = v8f_zero();
#pragma unroll
  for (int j = 0; j < 8; ++j) { mrow[j] = -3.0e38f; lrow[j] = 0.f; }

  const int kbmax = (q0 + 15) >> 5;  // last key block touching the diagonal
  for (int kb = 0; kb <= kbmax; ++kb) {
    const int k0 = kb * 32;
    v8f S[2];
#pragma unroll
    for (int nt = 0; nt < 2; ++nt) {
      S[nt] = v8f_zero();
#pragma unroll
      for (int kk = 0; kk < 2; ++kk) {
        v16bf bk;
        const bf16_t* src = kp + (size_t)(k0 + nt * 16 + ml) * HD + kk * 32 + 16 * sh;
        v8bf* p = (v8bf*)&bk;
        p[0] = *(const v8bf*)src;
        p[1] = *(const v8bf*)(src + 8);
        S[nt] = __builtin_amdgcn_wmma_f32_16x16x32_bf16(
            false, aq[kk], false, bk, (short)0, S[nt], false, false);
      }
    }
    // causal mask + online softmax (row = j + 8*sh, col = k0 + {ml, 16+ml})
    float fac[8];
#pragma unroll
    for (int j = 0; j < 8; ++j) {
      const int row = q0 + j + 8 * sh;
      float s0 = S[0][j];
      float s1 = S[1][j];
      if (k0 + ml > row)      s0 = -3.0e38f;
      if (k0 + 16 + ml > row) s1 = -3.0e38f;
      float mx = fmaxf(s0, s1);
      mx = fmaxf(mx, __shfl_xor(mx, 1));
      mx = fmaxf(mx, __shfl_xor(mx, 2));
      mx = fmaxf(mx, __shfl_xor(mx, 4));
      mx = fmaxf(mx, __shfl_xor(mx, 8));
      const float mn = fmaxf(mrow[j], mx);
      fac[j]  = __expf(mrow[j] - mn);
      mrow[j] = mn;
      const float p0 = __expf(s0 - mn);
      const float p1 = __expf(s1 - mn);
      float rs = p0 + p1;
      rs += __shfl_xor(rs, 1);
      rs += __shfl_xor(rs, 2);
      rs += __shfl_xor(rs, 4);
      rs += __shfl_xor(rs, 8);
      lrow[j] = lrow[j] * fac[j] + rs;
      Pl[wave][(j + 8 * sh) * PSTR + ml]      = (bf16_t)p0;
      Pl[wave][(j + 8 * sh) * PSTR + 16 + ml] = (bf16_t)p1;
    }
    // reload P as A fragment (per-wave LDS region; same-wave DS ops are ordered)
    v16bf ap;
    {
      v8bf* p = (v8bf*)&ap;
      const bf16_t* src = &Pl[wave][ml * PSTR];
      p[0] = *(const v8bf*)(src + 16 * sh);
      p[1] = *(const v8bf*)(src + 16 * sh + 8);
    }
#pragma unroll
    for (int nt = 0; nt < 4; ++nt) {
#pragma unroll
      for (int j = 0; j < 8; ++j) O[nt][j] *= fac[j];
      v16bf bv;  // column of V = row of V^T, contiguous over key index
      const bf16_t* src = vp + (size_t)(nt * 16 + ml) * SEQ + k0 + 16 * sh;
      v8bf* p = (v8bf*)&bv;
      p[0] = *(const v8bf*)src;
      p[1] = *(const v8bf*)(src + 8);
      O[nt] = __builtin_amdgcn_wmma_f32_16x16x32_bf16(
          false, ap, false, bv, (short)0, O[nt], false, false);
    }
  }

  const int b = bh >> 4, h = bh & 15;
#pragma unroll
  for (int nt = 0; nt < 4; ++nt) {
#pragma unroll
    for (int j = 0; j < 8; ++j) {
      const int t   = q0 + j + 8 * sh;
      const int row = b * SEQ + t;
      const int col = h * HD + nt * 16 + ml;
      attnb[(size_t)row * D_MODEL + col] = (bf16_t)(O[nt][j] / lrow[j]);
    }
  }
}

// ---------------------------------------------------------------- launcher
extern "C" void kernel_launch(void* const* d_in, const int* in_sizes, int n_in,
                              void* d_out, int out_size, void* d_ws, size_t ws_size,
                              hipStream_t stream) {
  const float* x    = (const float*)d_in[0];
  const float* Wqkv = (const float*)d_in[1];
  const float* Wout = (const float*)d_in[2];
  float* out = (float*)d_out;

  char* ws = (char*)d_ws;
  bf16_t* xb    = (bf16_t*)(ws + 0);               //  8 MiB  [4096,1024]
  bf16_t* wqkvb = (bf16_t*)(ws + 8388608);         //  6 MiB  [3072,1024]
  bf16_t* woutb = (bf16_t*)(ws + 14680064);        //  2 MiB  [1024,1024]
  bf16_t* qb    = (bf16_t*)(ws + 16777216);        //  8 MiB  [B,H,T,hd] (scaled)
  bf16_t* kbuf  = (bf16_t*)(ws + 25165824);        //  8 MiB  [B,H,T,hd]
  bf16_t* vtb   = (bf16_t*)(ws + 33554432);        //  8 MiB  [B,H,hd,T]
  bf16_t* attnb = (bf16_t*)(ws + 41943040);        //  8 MiB  [4096,1024]

  // 1) fp32 -> bf16 conversions
  cvt_f32_bf16<<<4096, 256, 0, stream>>>(x, xb, MTOT * D_MODEL);
  cvt_f32_bf16<<<3072, 256, 0, stream>>>(Wqkv, wqkvb, 3 * D_MODEL * D_MODEL);
  cvt_f32_bf16<<<1024, 256, 0, stream>>>(Wout, woutb, D_MODEL * D_MODEL);

  // 2) QKV projection: [4096,1024] x [3072,1024]^T -> scatter q/k/v^T
  gemm_wmma<1><<<dim3(24, 32), 256, 0, stream>>>(
      xb, wqkvb, nullptr, qb, kbuf, vtb, MTOT, 3 * D_MODEL, D_MODEL);

  // 3) fused causal flash attention
  flash_attn<<<BATCH * NHEAD * (SEQ / 64), 128, 0, stream>>>(qb, kbuf, vtb, attnb);

  // 4) output projection: [4096,1024] x [1024,1024]^T -> fp32 d_out
  gemm_wmma<0><<<dim3(8, 32), 256, 0, stream>>>(
      attnb, woutb, out, nullptr, nullptr, nullptr, MTOT, D_MODEL, D_MODEL);
}
